// LSTMPredictor_7249904796449
// MI455X (gfx1250) — compile-verified, matrix operand order checked by reading, not run on hardware
//
#include <hip/hip_runtime.h>

// ---------------------------------------------------------------------------
// CDNA5 (gfx1250) bi-LSTM. wave32. WMMA bf16 -> f32.
// ---------------------------------------------------------------------------

typedef __attribute__((ext_vector_type(16))) __bf16 v16bf;
typedef __attribute__((ext_vector_type(8)))  float  v8f;

union U16 {              // 32 bytes = one WMMA 16-bit operand (16 halves)
    uint4 u4[2];
    v16bf v;
};

__device__ __forceinline__ float sigmoidf_(float x) {
    return 1.0f / (1.0f + __expf(-x));
}

// ---------------------------------------------------------------------------
// f32 -> bf16 conversion (weights / activations)
// ---------------------------------------------------------------------------
__global__ void k_f32_to_bf16(const float* __restrict__ src,
                              __bf16* __restrict__ dst, int n) {
    int i = blockIdx.x * blockDim.x + threadIdx.x;
    if (i < n) dst[i] = (__bf16)src[i];
}

// ---------------------------------------------------------------------------
// Out[M,N] = A[M,K](bf16, row-major) @ W[N,K]^T (bf16, row-major) + bias1(+bias2)
// Block = 256 threads = 8 waves. grid = (M/16, N/128). Wave w -> n-tile 8*by+w.
// A tile (16 x K halves, K<=512) staged in LDS, shared by all 8 waves.
// K multiple of 32. M multiple of 16. N multiple of 128.
// ---------------------------------------------------------------------------
template <bool OUT_BF16>
__global__ __launch_bounds__(256) void
k_wmma_gemm_bias(const __bf16* __restrict__ A,
                 const __bf16* __restrict__ W,
                 const float* __restrict__ bias1,
                 const float* __restrict__ bias2,
                 void* __restrict__ Out,
                 int M, int N, int K) {
    __shared__ uint4 sA[1024];                     // 16 KB: 16 rows x 512 halves max

    const int tid  = threadIdx.x;
    const int wave = tid >> 5;
    const int lane = tid & 31;
    const int hl   = lane >> 4;                    // k-half selector (ISA A/B layout)
    const int r16  = lane & 15;

    const int mbase = blockIdx.x * 16;
    const int n0    = (blockIdx.y * 8 + wave) * 16;

    // ---- cooperative stage of the A tile into LDS (vectorized b128) ----
    const int ku4   = K >> 3;                      // uint4 per A row
    const int total = 16 * ku4;
    for (int q = tid; q < total; q += blockDim.x) {
        int row = q / ku4, cc = q - row * ku4;
        sA[q] = ((const uint4*)(A + (size_t)(mbase + row) * K))[cc];
    }
    __syncthreads();

    // ---- accumulator initialized with folded bias (broadcast over rows) ----
    v8f acc;
    {
        float b = bias1[n0 + r16] + (bias2 ? bias2[n0 + r16] : 0.0f);
#pragma unroll
        for (int v = 0; v < 8; ++v) acc[v] = b;
    }

    const uint4* wrow = (const uint4*)(W + (size_t)(n0 + r16) * K);

    for (int k0 = 0; k0 < K; k0 += 32) {
        const int c1 = (k0 + hl * 8) >> 3;         // first 8-half chunk
        const int c2 = (k0 + 16 + hl * 8) >> 3;    // second 8-half chunk
        U16 a, b;
        a.u4[0] = sA[r16 * ku4 + c1];              // A row = r16 (M index)
        a.u4[1] = sA[r16 * ku4 + c2];
        b.u4[0] = wrow[c1];                        // B col = r16 (N index)
        b.u4[1] = wrow[c2];
        acc = __builtin_amdgcn_wmma_f32_16x16x32_bf16(
                  false, a.v, false, b.v, (short)0, acc, false, false);
    }

    // ---- store: D element (m,n) = (v + 8*hl, r16) ----
#pragma unroll
    for (int v = 0; v < 8; ++v) {
        size_t idx = (size_t)(mbase + v + hl * 8) * N + n0 + r16;
        if (OUT_BF16) ((__bf16*)Out)[idx] = (__bf16)acc[v];
        else          ((float*)Out)[idx]  = acc[v];
    }
}

// ---------------------------------------------------------------------------
// Persistent LSTM recurrence. One block per (direction, batch-tile of 16 rows):
// grid.x = 2*8. Block = 512 threads = 16 waves. Wave w owns hidden columns
// [16w,16w+16) by computing gate n-tiles {w, w+16, w+32, w+48} (i,f,g,o), so
// cell state c stays in registers; h is exchanged through an 8 KB LDS buffer.
// w_hh [1024,256] bf16 streams from L2 every step (L2-resident, 512 KB).
// `zero` is always 0 but opaque to the compiler: it makes the weight address
// depend on the step index so LICM cannot hoist 256 VGPRs of loop-invariant
// weight tiles out of the 512-step loop (which spilled to scratch).
// last_only: forward dir stores only the final h; backward dir runs a single
// step (position T-1 is the first element of the reversed scan).
// ---------------------------------------------------------------------------
__global__ __launch_bounds__(512, 1) void
k_lstm_recurrent(const __bf16* __restrict__ xg_f,
                 const __bf16* __restrict__ xg_b,
                 const __bf16* __restrict__ whh_f,
                 const __bf16* __restrict__ whh_b,
                 __bf16* __restrict__ hseq,   // [B,T,512] or null
                 __bf16* __restrict__ hlast,  // [B,512]   or null
                 int T, int last_only, int zero) {
    constexpr int K = 256;                         // hidden size
    constexpr int G = 1024;                        // 4*H gate width
    __shared__ __bf16 hbuf[16 * K];                // 8 KB h tile (A operand)

    const int tid  = threadIdx.x;
    const int wave = tid >> 5;
    const int lane = tid & 31;
    const int hl   = lane >> 4;
    const int r16  = lane & 15;

    const int dir   = blockIdx.x >> 3;
    const int b0    = (blockIdx.x & 7) * 16;
    const bool rev  = (dir == 1);
    const int steps = (last_only && rev) ? 1 : T;

    const __bf16* xg  = rev ? xg_b  : xg_f;
    const __bf16* whh = rev ? whh_b : whh_f;
    const int col = wave * 16 + r16;               // owned hidden column

    // zero initial h
    for (int q = tid; q < (16 * K) / 8; q += blockDim.x)
        ((uint4*)hbuf)[q] = uint4{0u, 0u, 0u, 0u};
    v8f c = {};                                    // cell state in registers
    __syncthreads();

    for (int s = 0; s < steps; ++s) {
        const int t = rev ? (T - 1 - s) : s;

        // gate preactivations (C operand) from precomputed xg
        v8f acc[4];
#pragma unroll
        for (int g = 0; g < 4; ++g) {
            const int n0 = g * 256 + wave * 16;
#pragma unroll
            for (int v = 0; v < 8; ++v) {
                const int m = v + hl * 8;
                acc[g][v] = (float)xg[((size_t)(b0 + m) * T + t) * G + n0 + r16];
            }
        }

        // prefetch next step's gate preactivations (global_prefetch_b8)
        if (s + 1 < steps) {
            const int tn = rev ? (T - 2 - s) : (s + 1);
#pragma unroll
            for (int g = 0; g < 4; ++g) {
#pragma unroll
                for (int v = 0; v < 8; v += 4) {
                    const int m = v + hl * 8;
                    __builtin_prefetch(
                        &xg[((size_t)(b0 + m) * T + tn) * G + g * 256 + wave * 16 + r16],
                        0, 0);
                }
            }
        }

        // h @ W_hh^T : K=256 -> 8 k-chunks x 4 gate tiles of WMMA.
        // Weight base made step-dependent (zero*s == 0) to defeat LICM.
        const __bf16* wstep = whh + (size_t)zero * s;
        for (int k0 = 0; k0 < K; k0 += 32) {
            const int c1 = (k0 + hl * 8) >> 3;
            const int c2 = (k0 + 16 + hl * 8) >> 3;
            U16 a;
            a.u4[0] = ((const uint4*)hbuf)[r16 * (K / 8) + c1];
            a.u4[1] = ((const uint4*)hbuf)[r16 * (K / 8) + c2];
#pragma unroll
            for (int g = 0; g < 4; ++g) {
                const uint4* wr =
                    (const uint4*)(wstep + (size_t)(g * 256 + wave * 16 + r16) * K);
                U16 b;
                b.u4[0] = wr[c1];
                b.u4[1] = wr[c2];
                acc[g] = __builtin_amdgcn_wmma_f32_16x16x32_bf16(
                             false, a.v, false, b.v, (short)0, acc[g], false, false);
            }
        }
        __syncthreads();                           // everyone done reading old h

        // elementwise LSTM cell (i,f,g,o in PyTorch order)
#pragma unroll
        for (int v = 0; v < 8; ++v) {
            const float ig = sigmoidf_(acc[0][v]);
            const float fg = sigmoidf_(acc[1][v]);
            const float gg = tanhf(acc[2][v]);
            const float og = sigmoidf_(acc[3][v]);
            const float cn = fg * c[v] + ig * gg;
            c[v] = cn;
            const float hv = og * tanhf(cn);
            const int   m  = v + hl * 8;
            const __bf16 hb = (__bf16)hv;
            hbuf[m * K + col] = hb;
            if (!last_only)
                hseq[((size_t)(b0 + m) * T + t) * 512 + dir * 256 + col] = hb;
        }
        __syncthreads();                           // new h visible to all waves
    }

    if (last_only) {
#pragma unroll
        for (int v = 0; v < 8; ++v) {
            const int m = v + hl * 8;
            hlast[(size_t)(b0 + m) * 512 + dir * 256 + col] = hbuf[m * K + col];
        }
    }
}

// ---------------------------------------------------------------------------
// Host side
// ---------------------------------------------------------------------------
static inline int cdiv(int a, int b) { return (a + b - 1) / b; }

extern "C" void kernel_launch(void* const* d_in, const int* in_sizes, int n_in,
                              void* d_out, int out_size, void* d_ws, size_t ws_size,
                              hipStream_t stream) {
    (void)in_sizes; (void)n_in; (void)out_size; (void)ws_size;
    constexpr int B = 128, T = 512, D = 128, H = 256;
    constexpr int G = 4 * H;          // 1024
    constexpr int MT = B * T;         // 65536 rows for the big GEMMs

    const float* x        = (const float*)d_in[0];
    const float* wih0f    = (const float*)d_in[1];
    const float* whh0f    = (const float*)d_in[2];
    const float* bih0f    = (const float*)d_in[3];
    const float* bhh0f    = (const float*)d_in[4];
    const float* wih0b    = (const float*)d_in[5];
    const float* whh0b    = (const float*)d_in[6];
    const float* bih0b    = (const float*)d_in[7];
    const float* bhh0b    = (const float*)d_in[8];
    const float* wih1f    = (const float*)d_in[9];
    const float* whh1f    = (const float*)d_in[10];
    const float* bih1f    = (const float*)d_in[11];
    const float* bhh1f    = (const float*)d_in[12];
    const float* wih1b    = (const float*)d_in[13];
    const float* whh1b    = (const float*)d_in[14];
    const float* bih1b    = (const float*)d_in[15];
    const float* bhh1b    = (const float*)d_in[16];
    const float* fcw      = (const float*)d_in[17];
    const float* fcb      = (const float*)d_in[18];

    // ---- workspace carve-up (all bf16, every chunk 16B-aligned) ----
    __bf16* p = (__bf16*)d_ws;
    __bf16* xbf    = p; p += (size_t)B * T * D;        // 8388608
    __bf16* wih0fb = p; p += (size_t)G * D;            // 131072
    __bf16* wih0bb = p; p += (size_t)G * D;
    __bf16* whh0fb = p; p += (size_t)G * H;            // 262144
    __bf16* whh0bb = p; p += (size_t)G * H;
    __bf16* wih1fb = p; p += (size_t)G * 2 * H;        // 524288
    __bf16* wih1bb = p; p += (size_t)G * 2 * H;
    __bf16* whh1fb = p; p += (size_t)G * H;
    __bf16* whh1bb = p; p += (size_t)G * H;
    __bf16* fcwb   = p; p += (size_t)512 * 512;
    __bf16* xg0    = p; p += (size_t)MT * G;           // gate preacts, dir0 (reused L0/L1)
    __bf16* xg1    = p; p += (size_t)MT * G;           // gate preacts, dir1
    __bf16* h0seq  = p; p += (size_t)MT * 2 * H;       // layer-0 output sequence
    __bf16* h1last = p; p += (size_t)B * 2 * H;        // layer-1 h[:, -1, :]

    // ---- f32 -> bf16 conversions ----
    auto cvt = [&](const float* s, __bf16* d, int n) {
        k_f32_to_bf16<<<cdiv(n, 256), 256, 0, stream>>>(s, d, n);
    };
    cvt(x,     xbf,    B * T * D);
    cvt(wih0f, wih0fb, G * D);
    cvt(wih0b, wih0bb, G * D);
    cvt(whh0f, whh0fb, G * H);
    cvt(whh0b, whh0bb, G * H);
    cvt(wih1f, wih1fb, G * 2 * H);
    cvt(wih1b, wih1bb, G * 2 * H);
    cvt(whh1f, whh1fb, G * H);
    cvt(whh1b, whh1bb, G * H);
    cvt(fcw,   fcwb,   512 * 512);

    // ---- layer 0: input projections (bias b_ih + b_hh folded in) ----
    dim3 gBig(MT / 16, G / 128);
    k_wmma_gemm_bias<true><<<gBig, 256, 0, stream>>>(xbf, wih0fb, bih0f, bhh0f,
                                                     xg0, MT, G, D);
    k_wmma_gemm_bias<true><<<gBig, 256, 0, stream>>>(xbf, wih0bb, bih0b, bhh0b,
                                                     xg1, MT, G, D);

    // ---- layer 0: full bidirectional recurrence, store whole sequence ----
    k_lstm_recurrent<<<16, 512, 0, stream>>>(xg0, xg1, whh0fb, whh0bb,
                                             h0seq, nullptr, T, /*last_only=*/0, 0);

    // ---- layer 1: input projections over layer-0 sequence (K=512) ----
    k_wmma_gemm_bias<true><<<gBig, 256, 0, stream>>>(h0seq, wih1fb, bih1f, bhh1f,
                                                     xg0, MT, G, 2 * H);
    k_wmma_gemm_bias<true><<<gBig, 256, 0, stream>>>(h0seq, wih1bb, bih1b, bhh1b,
                                                     xg1, MT, G, 2 * H);

    // ---- layer 1: fwd = full scan keeping final h; bwd = single step ----
    k_lstm_recurrent<<<16, 512, 0, stream>>>(xg0, xg1, whh1fb, whh1bb,
                                             nullptr, h1last, T, /*last_only=*/1, 0);

    // ---- final FC: [128,512] @ fc_w^T + fc_b -> d_out (f32) ----
    k_wmma_gemm_bias<false><<<dim3(B / 16, 512 / 128), 256, 0, stream>>>(
        h1last, fcwb, fcb, nullptr, d_out, B, 512, 512);
}
